// MDRNN_basic_2989297238444
// MI455X (gfx1250) — matrix operand order repaired and checked by reading, not compiled, not run on hardware
//
#include <hip/hip_runtime.h>
#include <hip/hip_bf16.h>
#include <math.h>

// ---------------------------------------------------------------------------
// MD-LSTM (4-direction 2D LSTM) for MI455X / gfx1250, wave32 + WMMA.
//
// Critical-path-optimized version:
//   Gate pre-activations are split:  g = [h_up|x|1] @ Wrow  +  [h_left] @ Wleft
//   The Wrow term for cell j+1 is computed (7x K=32 WMMA) while cell j's
//   sequential part runs, and held in VGPRs; the recurrent chain per cell is
//   only:  ds_load(h_left) -> 7 independent v_wmma (C = pipelined regs)
//          -> gate staging -> elementwise LSTM update.
//   feat global loads are fully off the recurrent chain.
// ---------------------------------------------------------------------------

#define HID 20
#define GDIM 100      // 5*HID
#define GPAD 112      // 7 tiles of 16
#define NTILES 7
#define GRID 15
#define BT 16         // batch tile (WMMA M)

typedef _Float16 v16h __attribute__((ext_vector_type(16)));
typedef _Float16 v8h  __attribute__((ext_vector_type(8)));
typedef float    v8f  __attribute__((ext_vector_type(8)));

__device__ __forceinline__ float sigmoidf(float x) {
    return 1.0f / (1.0f + __expf(-x));
}

// ---------------- K1: conv3x3 + bias + maxpool2x2 + tanh -------------------
__global__ void k_conv_pool(const float* __restrict__ x,
                            const float* __restrict__ w,
                            const float* __restrict__ bias,
                            _Float16* __restrict__ feat, int total) {
    int idx = blockIdx.x * blockDim.x + threadIdx.x;
    if (idx >= total) return;
    int c  = idx % 10;
    int jj = (idx / 10) % GRID;
    int ii = (idx / (10 * GRID)) % GRID;
    int b  = idx / (10 * GRID * GRID);

    float mx = -1e30f;
#pragma unroll
    for (int di = 0; di < 2; ++di) {
#pragma unroll
        for (int dj = 0; dj < 2; ++dj) {
            int oi = 2 * ii + di;   // 0..29
            int oj = 2 * jj + dj;
            float s = bias[c];
#pragma unroll
            for (int ic = 0; ic < 3; ++ic)
#pragma unroll
                for (int kh = 0; kh < 3; ++kh)
#pragma unroll
                    for (int kw = 0; kw < 3; ++kw)
                        s += x[(((long)b * 3 + ic) * 32 + (oi + kh)) * 32 + (oj + kw)] *
                             w[((c * 3 + ic) * 3 + kh) * 3 + kw];
            mx = fmaxf(mx, s);
        }
    }
    feat[idx] = (_Float16)tanhf(mx);
}

// ---------------- K2: pack gate weight matrices ----------------------------
// Wrow : rows 0-19 = Whu, 20-29 = Wx, 30 = bg (constant-1 lane), 31 = 0
// Wleft: rows 0-19 = Whl, 20-31 = 0
// layout: (((d*7 + nt)*32 + k)*16 + n), f16
__global__ void k_pack_w(const float* __restrict__ Wx,
                         const float* __restrict__ Whu,
                         const float* __restrict__ Whl,
                         const float* __restrict__ bg,
                         _Float16* __restrict__ Wrow,
                         _Float16* __restrict__ Wleft, int total) {
    int idx = blockIdx.x * blockDim.x + threadIdx.x;
    if (idx >= total) return;
    int n  = idx & 15;
    int k  = (idx >> 4) & 31;
    int nt = (idx / (16 * 32)) % NTILES;
    int d  = idx / (16 * 32 * NTILES);
    int g  = nt * 16 + n;
    float vr = 0.0f, vl = 0.0f;
    if (g < GDIM) {
        if (k < 20) {
            vr = Whu[(d * 20 + k) * GDIM + g];
            vl = Whl[(d * 20 + k) * GDIM + g];
        } else if (k < 30) {
            vr = Wx[(d * 10 + (k - 20)) * GDIM + g];
        } else if (k == 30) {
            vr = bg[d * GDIM + g];
        }
    }
    Wrow[idx]  = (_Float16)vr;
    Wleft[idx] = (_Float16)vl;
}

// ---------------- K3: WMMA MD-LSTM scan (one wave per 16-batch x dir) ------
__global__ void __launch_bounds__(32)
k_scan(const _Float16* __restrict__ feat,
       const _Float16* __restrict__ WrowG,
       const _Float16* __restrict__ WleftG,
       float* __restrict__ feats_out) {
    const int wid = blockIdx.x;   // 0..511
    const int d   = wid & 3;      // direction
    const int b0  = (wid >> 2) * BT;
    const int l   = threadIdx.x;  // 0..31
    const int n   = l & 15;
    const int hi  = l >> 4;       // lane half

    const bool fh = (d == 1) || (d == 3);
    const bool fw = (d == 2) || (d == 3);

    __shared__ _Float16 hrow[GRID][BT][HID];  // wavefront hidden state
    __shared__ float    crow[GRID][BT][HID];  // wavefront cell state
    __shared__ _Float16 ubuf[BT][32];         // A staging: [h_up | x | 1 | 0]
    __shared__ _Float16 lbuf[BT][32];         // A staging: [h_left | 0]
    __shared__ float    gbuf[BT][GPAD];       // D staging (gate pre-acts)

    for (int t = l; t < GRID * BT * HID; t += 32) {
        ((_Float16*)hrow)[t] = (_Float16)0.0f;
        ((float*)crow)[t]    = 0.0f;
    }
    for (int t = l; t < BT * 32; t += 32) {
        ((_Float16*)ubuf)[t] = (_Float16)0.0f;
        ((_Float16*)lbuf)[t] = (_Float16)0.0f;
    }
    __syncthreads();
    if (l < BT) ubuf[l][30] = (_Float16)1.0f;  // bias lane (constant)
    __syncthreads();

    // Preload B fragments (k order matching A: element e <-> K row =
    // 8*hi + e for e<8, 16 + 8*hi + (e-8) for e>=8, column N = lane & 15).
    v16h Brow[NTILES], Bleft[NTILES];
#pragma unroll
    for (int nt = 0; nt < NTILES; ++nt) {
#pragma unroll
        for (int e = 0; e < 16; ++e) {
            int krow = (e < 8) ? (hi * 8 + e) : (16 + hi * 8 + (e - 8));
            size_t o  = ((size_t)(d * NTILES + nt) * 32 + krow) * 16 + n;
            Brow[nt][e]  = WrowG[o];
            Bleft[nt][e] = WleftG[o];
        }
    }

    // A-fragment gather from a [16][32] f16 LDS tile (ISA 16-bit 16x32 A
    // layout): lane row M = l&15; halves [8*hi..8*hi+7] and [16+8*hi..23+8*hi]
    // -> two ds_load_b128.
#define LOAD_A(buf, dst)                                                     \
    {                                                                        \
        const v8h* vr_ = (const v8h*)&buf[n][0];                             \
        v8h p0_ = vr_[hi], p1_ = vr_[hi + 2];                                \
        dst = __builtin_shufflevector(p0_, p1_, 0, 1, 2, 3, 4, 5, 6, 7, 8,   \
                                      9, 10, 11, 12, 13, 14, 15);            \
    }

    v8f gpre[NTILES];  // pipelined (h_up|x|1) @ Wrow contribution

    for (int i = 0; i < GRID; ++i) {
        const int ii = fh ? (GRID - 1 - i) : i;

        // ---- row prologue: zero h_left, stage ubuf for j=0 ----
        const int jj0 = fw ? (GRID - 1) : 0;
        for (int t = l; t < BT * HID; t += 32) {
            int m = t / HID, k = t % HID;
            lbuf[m][k] = (_Float16)0.0f;
            ubuf[m][k] = hrow[0][m][k];
        }
        for (int t = l; t < BT * 10; t += 32) {
            int m = t / 10, k = t % 10;
            ubuf[m][20 + k] =
                feat[(((size_t)(b0 + m) * GRID + ii) * GRID + jj0) * 10 + k];
        }
        __syncthreads();
        {
            v16h Au;
            LOAD_A(ubuf, Au);
#pragma unroll
            for (int nt = 0; nt < NTILES; ++nt) {
                v8f z = {};
                gpre[nt] = __builtin_amdgcn_wmma_f32_16x16x32_f16(
                    false, Au, false, Brow[nt], (short)0, z, false, false);
            }
        }

        for (int j = 0; j < GRID; ++j) {
            // ---- critical path: g = gpre + h_left @ Wleft ----
            v16h Al;
            LOAD_A(lbuf, Al);
#pragma unroll
            for (int nt = 0; nt < NTILES; ++nt) {
                v8f acc = __builtin_amdgcn_wmma_f32_16x16x32_f16(
                    false, Al, false, Bleft[nt], (short)0, gpre[nt], false,
                    false);
                // D layout: lane l, reg r -> M = r + 8*hi, N = l&15
#pragma unroll
                for (int r = 0; r < 8; ++r)
                    gbuf[r + hi * 8][nt * 16 + n] = acc[r];
            }

            // ---- off-chain: stage ubuf for next column (h_up_{j+1}, x) ----
            if (j < GRID - 1) {
                const int jn = fw ? (GRID - 2 - j) : (j + 1);
                for (int t = l; t < BT * HID; t += 32) {
                    int m = t / HID, k = t % HID;
                    ubuf[m][k] = hrow[j + 1][m][k];  // old value (h_up)
                }
                for (int t = l; t < BT * 10; t += 32) {
                    int m = t / 10, k = t % 10;
                    ubuf[m][20 + k] =
                        feat[(((size_t)(b0 + m) * GRID + ii) * GRID + jn) * 10 +
                             k];
                }
            }
            __syncthreads();

            // ---- elementwise LSTM update (writes h into lbuf for j+1) ----
            for (int t = l; t < BT * HID; t += 32) {
                int m = t / HID, k = t % HID;
                float gi = gbuf[m][k];
                float g1 = gbuf[m][20 + k];
                float g2 = gbuf[m][40 + k];
                float gz = gbuf[m][60 + k];
                float go = gbuf[m][80 + k];
                float I  = sigmoidf(gi);
                float F1 = sigmoidf(g1);
                float F2 = sigmoidf(g2);
                float O  = sigmoidf(go);
                float Z  = tanhf(gz);
                float cu = crow[j][m][k];                      // c_up (old)
                float cl = (j > 0) ? crow[j - 1][m][k] : 0.0f; // c_left (new)
                float c  = F1 * cu + F2 * cl + I * Z;
                float h  = O * tanhf(c);
                crow[j][m][k] = c;
                _Float16 hh   = (_Float16)h;
                hrow[j][m][k] = hh;
                lbuf[m][k]    = hh;   // becomes h_left of the next cell
            }
            __syncthreads();

            // ---- off-chain: pipelined Wrow WMMAs for next column ----
            if (j < GRID - 1) {
                v16h Au;
                LOAD_A(ubuf, Au);
#pragma unroll
                for (int nt = 0; nt < NTILES; ++nt) {
                    v8f z = {};
                    gpre[nt] = __builtin_amdgcn_wmma_f32_16x16x32_f16(
                        false, Au, false, Brow[nt], (short)0, z, false, false);
                }
            }
        }
    }

    // terminal cell's h is the only thing the reference consumes
    for (int t = l; t < BT * HID; t += 32) {
        int m = t / HID, k = t % HID;
        feats_out[(size_t)(b0 + m) * (4 * HID) + d * HID + k] =
            (float)hrow[GRID - 1][m][k];
    }
#undef LOAD_A
}

// ---------------- K4: final FC [B,80] x [80,10] ----------------------------
__global__ void k_fc(const float* __restrict__ feats,
                     const float* __restrict__ fw,
                     const float* __restrict__ fb,
                     float* __restrict__ out, int total) {
    int idx = blockIdx.x * blockDim.x + threadIdx.x;
    if (idx >= total) return;
    int o = idx % 10;
    int b = idx / 10;
    float s = fb[o];
#pragma unroll
    for (int k = 0; k < 4 * HID; ++k)
        s += feats[(size_t)b * (4 * HID) + k] * fw[k * 10 + o];
    out[idx] = s;
}

// ---------------------------------------------------------------------------
extern "C" void kernel_launch(void* const* d_in, const int* in_sizes, int n_in,
                              void* d_out, int out_size, void* d_ws, size_t ws_size,
                              hipStream_t stream) {
    const float* x      = (const float*)d_in[0];  // [2048,3,32,32]
    const float* conv_w = (const float*)d_in[1];  // [10,3,3,3]
    const float* conv_b = (const float*)d_in[2];  // [10]
    const float* Wx     = (const float*)d_in[3];  // [4,10,100]
    const float* Whu    = (const float*)d_in[4];  // [4,20,100]
    const float* Whl    = (const float*)d_in[5];  // [4,20,100]
    const float* bg     = (const float*)d_in[6];  // [4,100]
    const float* fc_w   = (const float*)d_in[7];  // [80,10]
    const float* fc_b   = (const float*)d_in[8];  // [10]
    float* out = (float*)d_out;                   // [2048,10]

    const int B = in_sizes[0] / (3 * 32 * 32);    // 2048

    // workspace layout
    char* ws = (char*)d_ws;
    _Float16* feat = (_Float16*)ws;                                // B*225*10 f16
    size_t feat_bytes = (size_t)B * GRID * GRID * 10 * sizeof(_Float16);
    size_t wpack_elems = (size_t)4 * NTILES * 32 * 16;
    _Float16* Wrow  = (_Float16*)(ws + feat_bytes);
    _Float16* Wleft = Wrow + wpack_elems;
    float* featsbuf = (float*)(ws + feat_bytes + 2 * wpack_elems * sizeof(_Float16));

    {
        int total = B * GRID * GRID * 10;
        k_conv_pool<<<(total + 255) / 256, 256, 0, stream>>>(x, conv_w, conv_b,
                                                             feat, total);
    }
    {
        int total = (int)wpack_elems;
        k_pack_w<<<(total + 255) / 256, 256, 0, stream>>>(Wx, Whu, Whl, bg,
                                                          Wrow, Wleft, total);
    }
    {
        int nwaves = (B / BT) * 4;  // 512 independent scans
        k_scan<<<nwaves, 32, 0, stream>>>(feat, Wrow, Wleft, featsbuf);
    }
    {
        int total = B * 10;
        k_fc<<<(total + 255) / 256, 256, 0, stream>>>(featsbuf, fc_w, fc_b,
                                                      out, total);
    }
}
